// EdgeQProj_68453188763807
// MI455X (gfx1250) — matrix operand order repaired and checked by reading, not compiled, not run on hardware
//
#include <hip/hip_runtime.h>
#include <hip/hip_bf16.h>

typedef __attribute__((ext_vector_type(16))) __bf16        v16bf;
typedef __attribute__((ext_vector_type(8)))  float         v8f;
typedef __attribute__((ext_vector_type(4)))  unsigned int  v4u;
typedef __attribute__((ext_vector_type(8)))  int           v8i;
typedef __attribute__((ext_vector_type(4)))  int           v4i;

#define WAVE 32

#if defined(__gfx1250__) && __has_builtin(__builtin_amdgcn_tensor_load_to_lds)
#define USE_TDM 1
#else
#define USE_TDM 0
#endif

// round-to-nearest-even f32 -> bf16 (as raw u16)
__device__ __forceinline__ unsigned short f2bf(float x) {
    unsigned int u = __float_as_uint(x);
    u += 0x7FFFu + ((u >> 16) & 1u);
    return (unsigned short)(u >> 16);
}

// ---- W [K,N] f32  ->  WT [N,K] bf16 (tiled transpose) ----
__global__ __launch_bounds__(256) void transpose_to_bf16(
    const float* __restrict__ W, unsigned short* __restrict__ WT, int K, int N)
{
    __shared__ unsigned short t[32][33];
    const int k0 = blockIdx.x * 32, n0 = blockIdx.y * 32;
    const int tx = threadIdx.x & 31, ty = threadIdx.x >> 5;   // 32 x 8
    #pragma unroll
    for (int i = 0; i < 32; i += 8)
        t[ty + i][tx] = f2bf(W[(size_t)(k0 + ty + i) * N + n0 + tx]);
    __syncthreads();
    #pragma unroll
    for (int i = 0; i < 32; i += 8)
        WT[(size_t)(n0 + ty + i) * K + k0 + tx] = t[tx][ty + i];
}

// C[M,N] = relu(A[M,K] @ W[K,N] + bias[N]).  A f32, W given transposed as bf16 WT[N,K].
// Block: 256 threads = 8 waves (WROWS x WCOLS), each wave owns a 32x32 output tile.
// B tile staged to LDS by the Tensor Data Mover (TENSORcnt) when available.
template<int BM, int BN, int WROWS, int WCOLS>
__global__ __launch_bounds__(256) void gemm_bias_relu_wmma(
    const float* __restrict__ A, const unsigned short* __restrict__ WT,
    const float* __restrict__ bias, float* __restrict__ C,
    int M, int N, int K)
{
    constexpr int BK  = 32;
    constexpr int LDA = BK + 8;               // bf16 elements (pad: spread LDS banks)
    constexpr int LDB = BK + 8;               // matches TDM pad: 16 DW interval + 4 DW pad
    constexpr unsigned SB_OFF = BM * LDA * 2; // byte offset of sB in dynamic LDS

    extern __shared__ char smem[];
    unsigned short* sA = (unsigned short*)smem;            // [BM][LDA]
    unsigned short* sB = (unsigned short*)(smem + SB_OFF); // [BN][LDB]

    const int tid   = threadIdx.x;
    const int wave  = tid / WAVE;
    const int lane  = tid % WAVE;
    const int wrow  = wave % WROWS;
    const int wcol  = wave / WROWS;
    const int tileM = blockIdx.x * BM;
    const int tileN = blockIdx.y * BN;
    const int hsel  = lane >> 4;
    const int l16   = lane & 15;

    v8f acc[2][2] = {};

    for (int k0 = 0; k0 < K; k0 += BK) {
#if USE_TDM
        // ---- B tile via Tensor Data Mover: WT rows [tileN, tileN+BN), cols [k0, k0+32) ----
        if (wave == 0) {
            const unsigned long long ga =
                (unsigned long long)(const void*)(WT + (size_t)tileN * K + k0);
            v4u g0;
            g0[0] = 1u;                                     // count=1 (valid), user mode
            g0[1] = SB_OFF;                                 // lds_addr (bytes)
            g0[2] = (unsigned)(ga & 0xFFFFFFFFull);
            g0[3] = (unsigned)((ga >> 32) & 0x01FFFFFFull) | (2u << 30); // type=2 (image)
            v8i g1;
            g1[0] = 0x06D10000;          // data_size=2B, pad_enable, interval=16DW, pad=4DW
            g1[1] = (int)((K & 0xFFFF) << 16);              // tensor_dim0 lo16
            g1[2] = (int)(((unsigned)K >> 16) | ((BN & 0xFFFF) << 16)); // dim0 hi | dim1 lo
            g1[3] = (int)((BK) << 16);                      // dim1 hi=0 | tile_dim0=32
            g1[4] = (int)BN;                                // tile_dim1=BN, tile_dim2=0
            g1[5] = (int)K;                                 // tensor_dim0_stride lo32 (elems)
            g1[6] = 0;                                      // stride hi | dim1_stride lo
            g1[7] = 0;
            v4i gz4 = {0, 0, 0, 0};
            v8i gz8 = {0, 0, 0, 0, 0, 0, 0, 0};
            __builtin_amdgcn_tensor_load_to_lds(g0, g1, gz4, gz4, gz8, 0);
        }
#else
        // ---- fallback: plain bf16 loads of WT tile (16B per thread-chunk) ----
        for (int e = tid; e < BN * (BK / 8); e += 256) {
            const int nn = e / (BK / 8);
            const int kk = (e % (BK / 8)) * 8;
            *(uint4*)&sB[nn * LDB + kk] =
                *(const uint4*)(WT + (size_t)(tileN + nn) * K + k0 + kk);
        }
#endif
        // ---- A tile: BM x 32 f32 -> bf16 (8 threads/row x float4) ----
        for (int r = tid >> 3; r < BM; r += 32) {
            const int kk = (tid & 7) * 4;
            const float4 v = *(const float4*)(A + (size_t)(tileM + r) * K + k0 + kk);
            uint2 p;
            p.x = (unsigned int)f2bf(v.x) | ((unsigned int)f2bf(v.y) << 16);
            p.y = (unsigned int)f2bf(v.z) | ((unsigned int)f2bf(v.w) << 16);
            *(uint2*)&sA[r * LDA + kk] = p;
        }
#if USE_TDM
        if (wave == 0) __builtin_amdgcn_s_wait_tensorcnt(0);
#endif
        __syncthreads();

        // ---- fragments per ISA 16-bit layouts, 2x2 WMMAs ----
        v16bf afrag[2], bfrag[2];
        #pragma unroll
        for (int i = 0; i < 2; ++i) {
            const unsigned short* p = &sA[(wrow * 32 + i * 16 + l16) * LDA];
            uint2* ap = (uint2*)&afrag[i];
            ap[0] = *(const uint2*)(p + hsel * 8);
            ap[1] = *(const uint2*)(p + hsel * 8 + 4);
            ap[2] = *(const uint2*)(p + 16 + hsel * 8);
            ap[3] = *(const uint2*)(p + 16 + hsel * 8 + 4);
            const unsigned short* q = &sB[(wcol * 32 + i * 16 + l16) * LDB];
            uint2* bp = (uint2*)&bfrag[i];
            bp[0] = *(const uint2*)(q + hsel * 8);
            bp[1] = *(const uint2*)(q + hsel * 8 + 4);
            bp[2] = *(const uint2*)(q + 16 + hsel * 8);
            bp[3] = *(const uint2*)(q + 16 + hsel * 8 + 4);
        }
        #pragma unroll
        for (int i = 0; i < 2; ++i)
            #pragma unroll
            for (int j = 0; j < 2; ++j)
                acc[i][j] = __builtin_amdgcn_wmma_f32_16x16x32_bf16(
                    false, afrag[i], false, bfrag[j],
                    (short)0, acc[i][j], false, false);
        __syncthreads();
    }

    // ---- epilogue: bias + relu; C/D layout: VGPR r -> M = r + 8*hsel, N = l16 ----
    #pragma unroll
    for (int i = 0; i < 2; ++i) {
        #pragma unroll
        for (int j = 0; j < 2; ++j) {
            const int nglob = tileN + wcol * 32 + j * 16 + l16;
            const float bn  = bias[nglob];
            const int mbase = tileM + wrow * 32 + i * 16 + hsel * 8;
            #pragma unroll
            for (int r = 0; r < 8; ++r) {
                float v = acc[i][j][r] + bn;
                v = v > 0.f ? v : 0.f;
                C[(size_t)(mbase + r) * N + nglob] = v;
            }
        }
    }
}

// norms[b][k] = || relu-q_h[b].reshape(512,8)[:,k] ||  (k = e % 8)
__global__ __launch_bounds__(256) void norm_kernel(const float* __restrict__ qh,
                                                   float* __restrict__ norms)
{
    __shared__ float red[256];
    const int b = blockIdx.x;
    float s = 0.f;
    for (int e = threadIdx.x; e < 4096; e += 256) {
        const float v = qh[(size_t)b * 4096 + e];
        s += v * v;
    }
    red[threadIdx.x] = s;
    __syncthreads();
    if (threadIdx.x < 8) {
        float t = 0.f;
        for (int j = threadIdx.x; j < 256; j += 8) t += red[j];
        norms[b * 8 + threadIdx.x] = sqrtf(t);
    }
}

// one wave per 4 edges: obj_e = node_h[b,src] + node_h[b,dst]; logit[k] = <obj_e, q_e[:,k]> / norm
__global__ __launch_bounds__(256) void edge_kernel(
    const int* __restrict__ indexes, const float* __restrict__ node_h,
    const float* __restrict__ qh, const float* __restrict__ norms,
    float* __restrict__ out, int nEdges)
{
    const int lane = threadIdx.x & 31;
    const int waveGlobal = (blockIdx.x * 256 + threadIdx.x) >> 5;
    #pragma unroll 1
    for (int e = 0; e < 4; ++e) {
        const int m = waveGlobal * 4 + e;
        if (m >= nEdges) return;
        const int idx = indexes[m];
        const int b   = idx >> 12;
        const int rem = idx & 4095;
        const int src = rem >> 6;
        const int dst = rem & 63;
        const float* r0 = node_h + ((size_t)b * 64 + src) * 512;
        const float* r1 = node_h + ((size_t)b * 64 + dst) * 512;
        const float* q  = qh + (size_t)b * 4096;

        float a0=0,a1=0,a2=0,a3=0,a4=0,a5=0,a6=0,a7=0;
        for (int h = lane; h < 512; h += 32) {
            const float o = r0[h] + r1[h];
            const float4 qa = *(const float4*)(q + h * 8);
            const float4 qb = *(const float4*)(q + h * 8 + 4);
            a0 += o * qa.x; a1 += o * qa.y; a2 += o * qa.z; a3 += o * qa.w;
            a4 += o * qb.x; a5 += o * qb.y; a6 += o * qb.z; a7 += o * qb.w;
        }
        #pragma unroll
        for (int off = 16; off > 0; off >>= 1) {
            a0 += __shfl_xor(a0, off, 32); a1 += __shfl_xor(a1, off, 32);
            a2 += __shfl_xor(a2, off, 32); a3 += __shfl_xor(a3, off, 32);
            a4 += __shfl_xor(a4, off, 32); a5 += __shfl_xor(a5, off, 32);
            a6 += __shfl_xor(a6, off, 32); a7 += __shfl_xor(a7, off, 32);
        }
        if (lane == 0) {
            const float* nm = norms + b * 8;
            float4 o1, o2;
            o1.x = a0 / nm[0]; o1.y = a1 / nm[1]; o1.z = a2 / nm[2]; o1.w = a3 / nm[3];
            o2.x = a4 / nm[4]; o2.y = a5 / nm[5]; o2.z = a6 / nm[6]; o2.w = a7 / nm[7];
            *(float4*)(out + (size_t)m * 8)     = o1;
            *(float4*)(out + (size_t)m * 8 + 4) = o2;
        }
    }
}

extern "C" void kernel_launch(void* const* d_in, const int* in_sizes, int n_in,
                              void* d_out, int out_size, void* d_ws, size_t ws_size,
                              hipStream_t stream) {
    const float* node_feats = (const float*)d_in[0];   // [64,64,2048]
    const float* q_feats    = (const float*)d_in[1];   // [64,1024]
    const int*   indexes    = (const int*)d_in[2];     // [32768]
    const float* W_obj      = (const float*)d_in[3];   // [2048,512]
    const float* b_obj      = (const float*)d_in[4];   // [512]
    const float* W_q        = (const float*)d_in[5];   // [1024,4096]
    const float* b_q        = (const float*)d_in[6];   // [4096]
    float* out = (float*)d_out;

    float* node_h = (float*)d_ws;                              // 4096*512 f32
    float* q_h    = node_h + (size_t)4096 * 512;               // 64*4096 f32
    float* norms  = q_h + (size_t)64 * 4096;                   // 64*8 f32
    unsigned short* WobjT = (unsigned short*)(norms + 512);    // [512][2048] bf16
    unsigned short* WqT   = WobjT + (size_t)512 * 2048;        // [4096][1024] bf16

    const int nEdges = in_sizes[2];
    const size_t shbytes = (size_t)(128 * 40 + 64 * 40) * 2;   // 15360 B, both configs

    // pre-transpose weights to bf16 [N][K]
    transpose_to_bf16<<<dim3(2048 / 32, 512 / 32), 256, 0, stream>>>(W_obj, WobjT, 2048, 512);
    transpose_to_bf16<<<dim3(1024 / 32, 4096 / 32), 256, 0, stream>>>(W_q, WqT, 1024, 4096);

    // GEMM1: [4096,2048] @ [2048,512]
    gemm_bias_relu_wmma<128, 64, 4, 2><<<dim3(32, 8), 256, shbytes, stream>>>(
        node_feats, WobjT, b_obj, node_h, 4096, 512, 2048);

    // GEMM2: [64,1024] @ [1024,4096]
    gemm_bias_relu_wmma<64, 128, 2, 4><<<dim3(1, 32), 256, shbytes, stream>>>(
        q_feats, WqT, b_q, q_h, 64, 4096, 1024);

    norm_kernel<<<64, 256, 0, stream>>>(q_h, norms);

    const int edgeBlocks = (nEdges + 31) / 32;
    edge_kernel<<<edgeBlocks, 256, 0, stream>>>(indexes, node_h, q_h, norms, out, nEdges);
}
